// CARCAttentionProcessor_72086731096322
// MI455X (gfx1250) — matrix operand order repaired and checked by reading, not compiled, not run on hardware
//
#include <hip/hip_runtime.h>
#include <stdint.h>

// ---------------------------------------------------------------------------
// Types for CDNA5 WMMA (wave32)
// ---------------------------------------------------------------------------
typedef __bf16 bf16;
typedef __attribute__((ext_vector_type(16))) __bf16 v16bf;
typedef __attribute__((ext_vector_type(8)))  __bf16 bf16x8;
typedef __attribute__((ext_vector_type(8)))  float  v8f;
typedef int v4i __attribute__((vector_size(16)));   // matches builtin prototype

union V16 { bf16x8 h8[2]; v16bf v; };

// Problem dims (fixed by the reference)
#define BB   4
#define LL   1024
#define DDIM 1280
#define HH   20
#define HDIM 64
#define BHN  80      // B*H
#define LKV  2048    // concat length
#define MM   4096    // B*L
#define ALPHA 0.48f

// ---------------------------------------------------------------------------
// CDNA5 async global->LDS copy (ASYNCcnt path), with safe fallback
// ---------------------------------------------------------------------------
#if defined(__has_builtin)
#  if __has_builtin(__builtin_amdgcn_global_load_async_to_lds_b128)
#    define USE_ASYNC_LDS 1
#  endif
#  if __has_builtin(__builtin_amdgcn_s_wait_asynccnt)
#    define HAVE_WAIT_ASYNC 1
#  endif
#endif

__device__ __forceinline__ void cp_b128(void* lds, const void* g) {
#ifdef USE_ASYNC_LDS
    __builtin_amdgcn_global_load_async_to_lds_b128(
        (__attribute__((address_space(1))) v4i*)(uintptr_t)g,
        (__attribute__((address_space(3))) v4i*)(uintptr_t)lds, 0, 0);
#else
    *(bf16x8*)lds = *(const bf16x8*)g;
#endif
}

template <int N>
__device__ __forceinline__ void wait_async() {
#ifdef USE_ASYNC_LDS
#  ifdef HAVE_WAIT_ASYNC
    __builtin_amdgcn_s_wait_asynccnt(N);
#  else
    asm volatile("s_wait_asynccnt %0" :: "i"(N) : "memory");
#  endif
#endif
}

// ---------------------------------------------------------------------------
// Pack / layout kernels (fp32 -> bf16, transposes, alpha-scaled KV bank)
// ---------------------------------------------------------------------------
__global__ void cast_f32_bf16(const float* __restrict__ in, bf16* __restrict__ out, int n) {
    int i = blockIdx.x * blockDim.x + threadIdx.x;
    if (i < n) out[i] = (bf16)in[i];
}

__global__ void transpose_cast(const float* __restrict__ W, bf16* __restrict__ Wt,
                               int K, int N) {
    int i = blockIdx.x * blockDim.x + threadIdx.x;   // output-linear
    if (i < K * N) {
        int n = i / K, k = i % K;
        Wt[i] = (bf16)W[(size_t)k * N + n];
    }
}

__global__ void pack_kbg(const float* __restrict__ Kbg, bf16* __restrict__ Kc, int n) {
    int i = blockIdx.x * blockDim.x + threadIdx.x;
    if (i < n) {
        int bh = i >> 16, rem = i & 65535, l = rem >> 6, d = rem & 63;
        Kc[((size_t)bh * LKV + LL + l) * HDIM + d] = (bf16)(ALPHA * Kbg[i]);
    }
}

__global__ void pack_vbg(const float* __restrict__ Vbg, bf16* __restrict__ Vt, int n) {
    int i = blockIdx.x * blockDim.x + threadIdx.x;
    if (i < n) {
        int bh = i >> 16, rem = i & 65535, l = rem >> 6, d = rem & 63;
        Vt[((size_t)bh * HDIM + d) * LKV + LL + l] = (bf16)(ALPHA * Vbg[i]);
    }
}

// ---------------------------------------------------------------------------
// WMMA GEMM with async-LDS double buffering.
// Block tile 128x64, 8 waves, each wave 32x32 (2x2 WMMA accumulators).
// mode: 0=Q head-major, 1=K concat, 2=V transposed, 3=fp32 out+bias
// ---------------------------------------------------------------------------
#define APAD 40   // 80B row pitch -> 20-bank stride, conflict-free, 16B aligned

__global__ __launch_bounds__(256) void gemm_bf16_wmma(
    const bf16* __restrict__ A, const bf16* __restrict__ Bt,
    void* __restrict__ Out, const float* __restrict__ bias, int mode)
{
    __shared__ __align__(16) bf16 Asm[2][128][APAD];
    __shared__ __align__(16) bf16 Bsm[2][64][APAD];

    const int tid  = threadIdx.x;
    const int lane = tid & 31;
    const int wid  = tid >> 5;
    const int h    = lane >> 4;
    const int ln   = lane & 15;
    const int mblk = blockIdx.y * 128;
    const int nblk = blockIdx.x * 64;
    const int wrow = (wid & 3) * 32;   // wave tile inside block
    const int wcol = (wid >> 2) * 32;

    v8f acc[2][2] = {};

    // cooperative staging: A 128x32 (2 chunks/thread) + B 64x32 (1 chunk/thread)
    auto stage = [&](int buf, int k0) {
        const int c1 = tid + 256;
        cp_b128(&Asm[buf][tid >> 2][(tid & 3) * 8],
                A + (size_t)(mblk + (tid >> 2)) * DDIM + k0 + (tid & 3) * 8);
        cp_b128(&Asm[buf][c1 >> 2][(c1 & 3) * 8],
                A + (size_t)(mblk + (c1 >> 2)) * DDIM + k0 + (c1 & 3) * 8);
        cp_b128(&Bsm[buf][tid >> 2][(tid & 3) * 8],
                Bt + (size_t)(nblk + (tid >> 2)) * DDIM + k0 + (tid & 3) * 8);
    };

    stage(0, 0);
    for (int k0 = 0; k0 < DDIM; k0 += 32) {
        const int buf = (k0 >> 5) & 1;
        if (k0 + 32 < DDIM) { stage(buf ^ 1, k0 + 32); wait_async<3>(); }
        else                { wait_async<0>(); }
        __syncthreads();

        V16 a[2], b[2];
#pragma unroll
        for (int t = 0; t < 2; ++t) {
            const int ar = wrow + t * 16 + ln;
            a[t].h8[0] = *(const bf16x8*)&Asm[buf][ar][h * 8];
            a[t].h8[1] = *(const bf16x8*)&Asm[buf][ar][16 + h * 8];
            const int br = wcol + t * 16 + ln;
            b[t].h8[0] = *(const bf16x8*)&Bsm[buf][br][h * 8];
            b[t].h8[1] = *(const bf16x8*)&Bsm[buf][br][16 + h * 8];
        }
#pragma unroll
        for (int mt = 0; mt < 2; ++mt)
#pragma unroll
            for (int nt = 0; nt < 2; ++nt)
                acc[mt][nt] = __builtin_amdgcn_wmma_f32_16x16x32_bf16(
                    false, a[mt].v, false, b[nt].v, (short)0, acc[mt][nt], false, false);
        __syncthreads();
    }

#pragma unroll
    for (int mt = 0; mt < 2; ++mt)
#pragma unroll
        for (int nt = 0; nt < 2; ++nt)
#pragma unroll
            for (int r = 0; r < 8; ++r) {
                const int m = mblk + wrow + mt * 16 + r + 8 * h;
                const int n = nblk + wcol + nt * 16 + ln;
                const float v = acc[mt][nt][r];
                if (mode == 3) {
                    ((float*)Out)[(size_t)m * DDIM + n] = v + bias[n];
                } else {
                    const int b_ = m >> 10, l = m & 1023;
                    const int hh = n >> 6,  d = n & 63;
                    const size_t bh = (size_t)b_ * HH + hh;
                    bf16* o = (bf16*)Out;
                    if (mode == 0)       o[(bh * LL  + l) * HDIM + d] = (bf16)v;
                    else if (mode == 1)  o[(bh * LKV + l) * HDIM + d] = (bf16)v;
                    else                 o[(bh * HDIM + d) * LKV + l] = (bf16)v;
                }
            }
}

// ---------------------------------------------------------------------------
// Flash attention: 8 waves/block share ONE head; K/V chunks staged once per
// block via async-LDS double buffering (8x less L2 traffic than per-wave).
// ---------------------------------------------------------------------------
__global__ __launch_bounds__(256) void attn_fa_wmma(
    const bf16* __restrict__ Qb,   // [80][1024][64]
    const bf16* __restrict__ Kc,   // [80][2048][64]
    const bf16* __restrict__ Vt,   // [80][64][2048]
    bf16* __restrict__ ctxb)       // [4096][1280]
{
    __shared__ __align__(16) bf16 Ks[2][32][72];   // [buf][key][d], padded
    __shared__ __align__(16) bf16 Vs[2][64][APAD]; // [buf][d][key], padded
    __shared__ __align__(16) bf16 Pa[8][16][APAD]; // per-wave P relayout

    const int tid  = threadIdx.x;
    const int lane = tid & 31;
    const int wid  = tid >> 5;
    const int h    = lane >> 4;
    const int ln   = lane & 15;
    const int bh   = blockIdx.x >> 3;                       // one head per block
    const int q0   = ((blockIdx.x & 7) * 8 + wid) * 16;     // wave's query tile
    const float scale = 0.125f;                             // 1/sqrt(64)

    const bf16* kbase = Kc + (size_t)bh * LKV * HDIM;
    const bf16* vbase = Vt + (size_t)bh * HDIM * LKV;

    // Q fragments (K-dim 64), loop invariant
    const bf16* qrow = Qb + ((size_t)bh * LL + q0 + ln) * HDIM;
    V16 aq[2];
#pragma unroll
    for (int t = 0; t < 2; ++t) {
        aq[t].h8[0] = *(const bf16x8*)(qrow + t * 32 +      h * 8);
        aq[t].h8[1] = *(const bf16x8*)(qrow + t * 32 + 16 + h * 8);
    }

    float mrun[8], lrun[8];
#pragma unroll
    for (int r = 0; r < 8; ++r) { mrun[r] = -3.0e38f; lrun[r] = 0.0f; }
    v8f acc[4] = {};

    // staging: K chunk 32x64 (1 chunk/thread), V chunk 64x32 (1 chunk/thread)
    auto stage = [&](int buf, int j0) {
        cp_b128(&Ks[buf][tid >> 3][(tid & 7) * 8],
                kbase + (size_t)(j0 + (tid >> 3)) * HDIM + (tid & 7) * 8);
        cp_b128(&Vs[buf][tid >> 2][(tid & 3) * 8],
                vbase + (size_t)(tid >> 2) * LKV + j0 + (tid & 3) * 8);
    };

    stage(0, 0);
    for (int j0 = 0; j0 < LKV; j0 += 32) {
        const int buf = (j0 >> 5) & 1;
        if (j0 + 32 < LKV) { stage(buf ^ 1, j0 + 32); wait_async<2>(); }
        else               { wait_async<0>(); }
        __syncthreads();

        // ---- S = Q*K^T for 32 keys (4 WMMA) ----
        v8f s[2];
#pragma unroll
        for (int t = 0; t < 2; ++t) {
            const int kr = t * 16 + ln;
            V16 b0, b1;
            b0.h8[0] = *(const bf16x8*)&Ks[buf][kr][h * 8];
            b0.h8[1] = *(const bf16x8*)&Ks[buf][kr][16 + h * 8];
            b1.h8[0] = *(const bf16x8*)&Ks[buf][kr][32 + h * 8];
            b1.h8[1] = *(const bf16x8*)&Ks[buf][kr][32 + 16 + h * 8];
            v8f z = {};
            z    = __builtin_amdgcn_wmma_f32_16x16x32_bf16(false, aq[0].v, false, b0.v, (short)0, z, false, false);
            s[t] = __builtin_amdgcn_wmma_f32_16x16x32_bf16(false, aq[1].v, false, b1.v, (short)0, z, false, false);
        }

        // ---- online softmax (row stats per register r in 16-lane groups) ----
        float corr[8], p0[8], p1[8];
#pragma unroll
        for (int r = 0; r < 8; ++r) {
            float v0 = s[0][r] * scale, v1 = s[1][r] * scale;
            float cm = fmaxf(v0, v1);
#pragma unroll
            for (int mk = 1; mk < 16; mk <<= 1) cm = fmaxf(cm, __shfl_xor(cm, mk, 32));
            float nm = fmaxf(mrun[r], cm);
            corr[r] = __expf(mrun[r] - nm);
            p0[r]   = __expf(v0 - nm);
            p1[r]   = __expf(v1 - nm);
            float rs = p0[r] + p1[r];
#pragma unroll
            for (int mk = 1; mk < 16; mk <<= 1) rs += __shfl_xor(rs, mk, 32);
            lrun[r] = lrun[r] * corr[r] + rs;
            mrun[r] = nm;
        }
#pragma unroll
        for (int dt = 0; dt < 4; ++dt)
#pragma unroll
            for (int r = 0; r < 8; ++r) acc[dt][r] *= corr[r];

        // ---- C-layout -> A-layout for P via per-wave LDS ----
#pragma unroll
        for (int r = 0; r < 8; ++r) {
            Pa[wid][r + 8 * h][ln]      = (bf16)p0[r];
            Pa[wid][r + 8 * h][16 + ln] = (bf16)p1[r];
        }
        V16 ap;
        ap.h8[0] = *(const bf16x8*)&Pa[wid][ln][h * 8];
        ap.h8[1] = *(const bf16x8*)&Pa[wid][ln][16 + h * 8];

        // ---- ctx += P * V (4 WMMA) ----
#pragma unroll
        for (int dt = 0; dt < 4; ++dt) {
            const int vr = dt * 16 + ln;
            V16 bv;
            bv.h8[0] = *(const bf16x8*)&Vs[buf][vr][h * 8];
            bv.h8[1] = *(const bf16x8*)&Vs[buf][vr][16 + h * 8];
            acc[dt] = __builtin_amdgcn_wmma_f32_16x16x32_bf16(
                false, ap.v, false, bv.v, (short)0, acc[dt], false, false);
        }
        __syncthreads();
    }

    // ---- epilogue: normalize, scatter to batch-to-head [4096][1280] ----
    const int b_ = bh / HH, hh = bh % HH;
#pragma unroll
    for (int dt = 0; dt < 4; ++dt)
#pragma unroll
        for (int r = 0; r < 8; ++r) {
            const int m = b_ * LL + q0 + r + 8 * h;
            const int n = hh * HDIM + dt * 16 + ln;
            ctxb[(size_t)m * DDIM + n] = (bf16)(acc[dt][r] / lrun[r]);
        }
}

// ---------------------------------------------------------------------------
// Launch
// ---------------------------------------------------------------------------
extern "C" void kernel_launch(void* const* d_in, const int* in_sizes, int n_in,
                              void* d_out, int out_size, void* d_ws, size_t ws_size,
                              hipStream_t stream) {
    const float* hs  = (const float*)d_in[0];
    const float* Kbg = (const float*)d_in[1];
    const float* Vbg = (const float*)d_in[2];
    const float* Wq  = (const float*)d_in[3];
    const float* Wk  = (const float*)d_in[4];
    const float* Wv  = (const float*)d_in[5];
    const float* Wo  = (const float*)d_in[6];
    const float* bo  = (const float*)d_in[7];
    float* out = (float*)d_out;

    char* ws = (char*)d_ws;
    size_t off = 0;
    auto carve = [&](size_t elems) {
        void* p = ws + off;
        off = (off + elems * sizeof(bf16) + 255) & ~(size_t)255;
        return (bf16*)p;
    };
    bf16* hsb  = carve((size_t)MM * DDIM);
    bf16* WqT  = carve((size_t)DDIM * DDIM);
    bf16* WkT  = carve((size_t)DDIM * DDIM);
    bf16* WvT  = carve((size_t)DDIM * DDIM);
    bf16* WoT  = carve((size_t)DDIM * DDIM);
    bf16* Qb   = carve((size_t)BHN * LL * HDIM);
    bf16* Kc   = carve((size_t)BHN * LKV * HDIM);
    bf16* Vt   = carve((size_t)BHN * HDIM * LKV);
    bf16* ctxb = carve((size_t)MM * DDIM);

    const int nHS = MM * DDIM, nW = DDIM * DDIM, nKV = BHN * LL * HDIM;
    const int T = 256;

    cast_f32_bf16 <<<(nHS + T - 1) / T, T, 0, stream>>>(hs, hsb, nHS);
    transpose_cast<<<(nW  + T - 1) / T, T, 0, stream>>>(Wq, WqT, DDIM, DDIM);
    transpose_cast<<<(nW  + T - 1) / T, T, 0, stream>>>(Wk, WkT, DDIM, DDIM);
    transpose_cast<<<(nW  + T - 1) / T, T, 0, stream>>>(Wv, WvT, DDIM, DDIM);
    transpose_cast<<<(nW  + T - 1) / T, T, 0, stream>>>(Wo, WoT, DDIM, DDIM);
    pack_kbg      <<<(nKV + T - 1) / T, T, 0, stream>>>(Kbg, Kc, nKV);
    pack_vbg      <<<(nKV + T - 1) / T, T, 0, stream>>>(Vbg, Vt, nKV);

    dim3 gg(DDIM / 64, MM / 128);   // (20, 32)
    gemm_bf16_wmma<<<gg, T, 0, stream>>>(hsb, WqT, (void*)Qb, nullptr, 0);
    gemm_bf16_wmma<<<gg, T, 0, stream>>>(hsb, WkT, (void*)Kc, nullptr, 1);
    gemm_bf16_wmma<<<gg, T, 0, stream>>>(hsb, WvT, (void*)Vt, nullptr, 2);

    attn_fa_wmma<<<(BHN * (LL / 16)) / 8, T, 0, stream>>>(Qb, Kc, Vt, ctxb);

    gemm_bf16_wmma<<<gg, T, 0, stream>>>(ctxb, WoT, (void*)out, bo, 3);
}
